// FrameEventTrackingNet_42649025249454
// MI455X (gfx1250) — matrix operand order repaired and checked by reading, not compiled, not run on hardware
//
#include <hip/hip_runtime.h>
#include <hip/hip_bf16.h>
#include <math.h>

typedef __attribute__((ext_vector_type(16))) _Float16 v16h;
typedef __attribute__((ext_vector_type(8)))  _Float16 v8h;
typedef __attribute__((ext_vector_type(8)))  float    v8f;

// ---- problem constants (from reference) ----
constexpr int T_ = 8, B_ = 2, N_ = 256, C_ = 128, H_ = 96, W_ = 96;
constexpr int P_ = 7, ITERS_ = 4, EMB_ = 64;
constexpr int CIN_ = 392, COUT_ = 130;
constexpr int BN_ = B_ * N_;            // 512 instances
constexpr int NJ_ = BN_ * T_;           // 4096 = GEMM "N" dimension (bn*8 + t)
constexpr int CP_ = 416;                // padded channel stride (13*32)
constexpr int KP_ = 416;                // padded K
constexpr int KB_ = KP_ / 32;           // 13 WMMA K-chunks
constexpr int MP_MAIN = 400;            // 392 -> 25 tiles of 16
constexpr int MP_OUT  = 144;            // 130 -> 9 tiles of 16
constexpr int NT_ = 4;                  // N-tiles per wave (A-fragment reuse)
constexpr size_t XSLOT_ = (size_t)KP_ * NJ_;  // halfs per staged activation slot
constexpr float EPS_ = 1e-5f;

__device__ __forceinline__ float gelu_exact(float x) {
  return 0.5f * x * (1.0f + erff(x * 0.70710678118654752f));
}

// ---------------- state init ----------------
__global__ void k_init_traj(const float* __restrict__ tin, float* __restrict__ traj) {
  int idx = blockIdx.x * blockDim.x + threadIdx.x;     // T*B*N
  if (idx >= T_ * B_ * N_) return;
  int t = idx / (B_ * N_); int r = idx % (B_ * N_);
  int b = r / N_; int n = r % N_;
  int tb = t * B_ + b;
  // traj_init: [T,B,2,N], state traj: [T,B,N,2] = traj_init/SCALE transposed
  traj[((size_t)tb * N_ + n) * 2 + 0] = tin[((size_t)tb * 2 + 0) * N_ + n] * 0.25f;
  traj[((size_t)tb * N_ + n) * 2 + 1] = tin[((size_t)tb * 2 + 1) * N_ + n] * 0.25f;
}

__global__ void k_init_qv(const float* __restrict__ qin, float* __restrict__ qv) {
  int idx = blockIdx.x * blockDim.x + threadIdx.x;     // T*B*N*C
  if (idx >= T_ * B_ * N_ * C_) return;
  int c = idx % C_; int r = idx / C_;
  int n = r % N_; r /= N_;
  int b = r % B_; int t = r / B_;
  // qv state [T,B,N,C] = broadcast of query_vectors_init [B,C,N] transposed
  qv[(((size_t)(t * B_ + b) * N_) + n) * C_ + c] = qin[((size_t)b * C_ + c) * N_ + n];
}

// ---------------- pyramid (avg pool k=2 s=2) ----------------
__global__ void k_pool(const float* __restrict__ in, float* __restrict__ out,
                       int Hin, int Win) {
  int Ho = Hin >> 1, Wo = Win >> 1;
  long long total = (long long)T_ * B_ * C_ * Ho * Wo;
  long long id = (long long)blockIdx.x * blockDim.x + threadIdx.x;
  if (id >= total) return;
  int wo = (int)(id % Wo); long long r = id / Wo;
  int ho = (int)(r % Ho);  long long tc = r / Ho;   // [0, T*B*C)
  const float* p = in + (tc * Hin + (long long)ho * 2) * Win + (long long)wo * 2;
  out[id] = 0.25f * (p[0] + p[1] + p[Win] + p[Win + 1]);
}

// ---------------- weight f32 -> f16 staging (padded, [KS][MP][KP]) ----------------
__global__ void k_wcast(const float* __restrict__ w, _Float16* __restrict__ dst,
                        int Mo, int Ki, int KS, int MP) {
  int tot = KS * MP * KP_;
  int id = blockIdx.x * blockDim.x + threadIdx.x;
  if (id >= tot) return;
  int s = id / (MP * KP_); int r = id % (MP * KP_);
  int m = r / KP_; int k = r % KP_;
  float v = 0.f;
  if (m < Mo && k < Ki) v = w[((size_t)m * Ki + k) * KS + s];
  dst[((size_t)s * MP + m) * KP_ + k] = (_Float16)v;
}

// ---- activation stage: f32 [j][CP_] -> f16 K-major [KP_][NJ_], time-shifted + 0-pad ----
// xh[k][j] = x[k][bn, t+shift]  (zero outside t-range / K-pad). shift=0 for 1x1 convs;
// shift=-2..2 builds the 5 pre-shifted copies that turn the k=5 conv into 5 clean GEMM taps.
__global__ void k_stage(const float* __restrict__ act, _Float16* __restrict__ xh, int shift) {
  int id = blockIdx.x * blockDim.x + threadIdx.x;
  if (id >= KP_ * NJ_) return;
  int k = id / NJ_, j = id % NJ_;
  int ts = (j & 7) + shift;
  float v = 0.f;
  if (k < CIN_ && ts >= 0 && ts < 8) {
    int jsrc = (j & ~7) | ts;
    v = act[(size_t)jsrc * CP_ + k];
  }
  xh[id] = (_Float16)v;
}

// ---------------- WMMA GEMM ----------------
// out[j][m] = bias[m] + sum_{s,k} W[s][m][k] * Xs[s][k][j]
// xh: [KS][KP_][NJ_] f16 (pre-shifted slots).  wh: [KS][MP][KP_] f16.  out: [NJ_][CP_] f32.
// One wave32 per 16x(16*NT_) output tile; A fragment reused across NT_ B fragments.
__global__ void k_gemm_wmma(const _Float16* __restrict__ xh,
                            const _Float16* __restrict__ wh,
                            const float* __restrict__ bias,
                            float* __restrict__ out,
                            int Mreal, int Mtiles, int MP, int KS) {
  const int ngrp = NJ_ / (16 * NT_);          // 64 column groups
  int wave = blockIdx.x * (blockDim.x >> 5) + (threadIdx.x >> 5);
  if (wave >= Mtiles * ngrp) return;          // uniform per wave: EXEC stays all-ones
  int mt = wave % Mtiles;
  int ng = wave / Mtiles;
  int lane = threadIdx.x & 31;
  int lm = lane & 15;
  int hi = lane >> 4;

  v8f acc0 = {}, acc1 = {}, acc2 = {}, acc3 = {};
  const int klane = lm + hi * 16;             // unused; B fragment maps lane -> N column, K -> element
  for (int s = 0; s < KS; ++s) {
    const _Float16* xbase = xh + (size_t)s * XSLOT_ + (size_t)ng * (16 * NT_);
    const _Float16* wrow  = wh + ((size_t)s * MP + (mt * 16 + lm)) * KP_;  // A row = M
    for (int kb = 0; kb < KB_; ++kb) {
      // A: per ISA layout, each half-fragment is 8 contiguous halfs -> two 16B loads
      v8h alo = *(const v8h*)(wrow + kb * 32 + hi * 8);
      v8h ahi = *(const v8h*)(wrow + kb * 32 + 16 + hi * 8);
      v16h a = __builtin_shufflevector(alo, ahi, 0, 1, 2, 3, 4, 5, 6, 7,
                                       8, 9, 10, 11, 12, 13, 14, 15);
      // B: lane <-> N column, 16 K values in table order (strided gather)
      const _Float16* xr = xbase + (size_t)(kb * 32 + hi * 8) * NJ_ + lm;
      v16h b0, b1, b2, b3;
#pragma unroll
      for (int i = 0; i < 16; ++i) {
        size_t ko = (size_t)((i & 7) + (i >> 3) * 16) * NJ_;
        b0[i] = xr[ko + 0];
        b1[i] = xr[ko + 16];
        b2[i] = xr[ko + 32];
        b3[i] = xr[ko + 48];
      }
      acc0 = __builtin_amdgcn_wmma_f32_16x16x32_f16(false, a, false, b0, (short)0, acc0, false, false);
      acc1 = __builtin_amdgcn_wmma_f32_16x16x32_f16(false, a, false, b1, (short)0, acc1, false, false);
      acc2 = __builtin_amdgcn_wmma_f32_16x16x32_f16(false, a, false, b2, (short)0, acc2, false, false);
      acc3 = __builtin_amdgcn_wmma_f32_16x16x32_f16(false, a, false, b3, (short)0, acc3, false, false);
    }
  }
  // D layout: lane -> N column, vgpr v -> M = v + 8*hi
  v8f accs[NT_] = {acc0, acc1, acc2, acc3};
#pragma unroll
  for (int u = 0; u < NT_; ++u) {
    int j = ng * (16 * NT_) + u * 16 + lm;
    float* orow = out + (size_t)j * CP_;
#pragma unroll
    for (int v = 0; v < 8; ++v) {
      int m = mt * 16 + v + hi * 8;
      if (m < Mreal) orow[m] = accs[u][v] + bias[m];
    }
  }
}

// ---------------- instance norm over T (+optional residual) + exact GELU ----------------
__global__ void k_inorm_gelu(const float* __restrict__ y,
                             const float* __restrict__ res,   // may be nullptr
                             float* __restrict__ dst) {
  int id = blockIdx.x * blockDim.x + threadIdx.x;
  if (id >= BN_ * CIN_) return;
  int bn = id / CIN_, c = id % CIN_;
  size_t base = (size_t)bn * T_ * CP_ + c;
  float m = 0.f, m2 = 0.f;
#pragma unroll
  for (int t = 0; t < T_; ++t) {
    float v = y[base + (size_t)t * CP_];
    m += v; m2 += v * v;
  }
  m *= 0.125f;
  m2 = m2 * 0.125f - m * m;
  float rs = rsqrtf(m2 + EPS_);
#pragma unroll
  for (int t = 0; t < T_; ++t) {
    size_t o = base + (size_t)t * CP_;
    float v = (y[o] - m) * rs;
    if (res) v += res[o];
    dst[o] = gelu_exact(v);
  }
}

// ---------------- depthwise temporal conv k=5 pad=2 ----------------
__global__ void k_dw(const float* __restrict__ x, const float* __restrict__ w,
                     const float* __restrict__ b, float* __restrict__ y) {
  int id = blockIdx.x * blockDim.x + threadIdx.x;
  if (id >= BN_ * CIN_) return;
  int bn = id / CIN_, c = id % CIN_;
  size_t base = (size_t)bn * T_ * CP_ + c;
  float wv[5];
#pragma unroll
  for (int s = 0; s < 5; ++s) wv[s] = w[c * 5 + s];
  float bb = b[c];
  float xin[T_];
#pragma unroll
  for (int t = 0; t < T_; ++t) xin[t] = x[base + (size_t)t * CP_];
#pragma unroll
  for (int t = 0; t < T_; ++t) {
    float s = bb;
#pragma unroll
    for (int k = 0; k < 5; ++k) {
      int ts = t + k - 2;
      if (ts >= 0 && ts < T_) s += wv[k] * xin[ts];
    }
    y[base + (size_t)t * CP_] = s;
  }
}

// ---------------- build base channels: [traj(2) | flow(2) | emb(64) | qv(128)] ----------------
__global__ void k_build(const float* __restrict__ traj, const float* __restrict__ qv,
                        float* __restrict__ act) {
  int idx = blockIdx.x * blockDim.x + threadIdx.x;
  if (idx >= T_ * B_ * N_) return;
  int t = idx / (B_ * N_); int r = idx % (B_ * N_);
  int b = r / N_; int n = r % N_;
  int tb = t * B_ + b;
  size_t j = (size_t)(b * N_ + n) * T_ + t;
  float* row = act + j * CP_;
  float tx  = traj[((size_t)tb * N_ + n) * 2 + 0];
  float ty  = traj[((size_t)tb * N_ + n) * 2 + 1];
  float t0x = traj[((size_t)b * N_ + n) * 2 + 0];   // t=0 frame
  float t0y = traj[((size_t)b * N_ + n) * 2 + 1];
  float fx = tx - t0x, fy = ty - t0y;
  row[0] = tx; row[1] = ty; row[2] = fx; row[3] = fy;
#pragma unroll
  for (int i = 0; i < 16; ++i) {
    float om = powf(10000.f, -(float)i / 15.f);
    row[4 + i]  = sinf(fx * om);
    row[20 + i] = cosf(fx * om);
    row[36 + i] = sinf(fy * om);
    row[52 + i] = cosf(fy * om);
  }
  const float* qp = qv + ((size_t)tb * N_ + n) * C_;
  for (int c = 0; c < C_; ++c) row[68 + c] = qp[c];
}

// ---------------- bilinear 7x7 patch correlation for one pyramid level ----------------
// grid = T*B*N blocks, 128 threads (1 thread = 1 channel); wave32 shfl + LDS reduce.
__global__ void k_corr(const float* __restrict__ feat, int Hl, int Wl,
                       const float* __restrict__ traj, const float* __restrict__ qv,
                       float* __restrict__ act, float inv, int col0) {
  __shared__ float red[4];
  int idx = blockIdx.x;
  int t = idx / (B_ * N_); int r = idx % (B_ * N_);
  int b = r / N_; int n = r % N_;
  int tid = threadIdx.x;                       // channel
  int tb = t * B_ + b;
  float tx = traj[((size_t)tb * N_ + n) * 2 + 0] * inv;
  float ty = traj[((size_t)tb * N_ + n) * 2 + 1] * inv;
  float qc = qv[((size_t)tb * N_ + n) * C_ + tid];
  const float* f = feat + ((size_t)tb * C_ + tid) * Hl * Wl;
  size_t jrow = ((size_t)(b * N_ + n) * T_ + t) * CP_ + col0;
  for (int py = 0; py < P_; ++py) {
    for (int px = 0; px < P_; ++px) {
      float gx = tx + (float)(px - 3);
      float gy = ty + (float)(py - 3);
      float x0 = floorf(gx), y0 = floorf(gy);
      float wx1 = gx - x0, wy1 = gy - y0;
      int xi = (int)x0, yi = (int)y0;
      float s = 0.f;
#pragma unroll
      for (int dy = 0; dy < 2; ++dy)
#pragma unroll
        for (int dx = 0; dx < 2; ++dx) {
          int xc = xi + dx, yc = yi + dy;
          float w = (dx ? wx1 : 1.f - wx1) * (dy ? wy1 : 1.f - wy1);
          if (xc >= 0 && xc < Wl && yc >= 0 && yc < Hl)
            s += f[(size_t)yc * Wl + xc] * w;
        }
      float v = s * qc;
#pragma unroll
      for (int off = 16; off > 0; off >>= 1) v += __shfl_xor(v, off, 32);
      if ((tid & 31) == 0) red[tid >> 5] = v;
      __syncthreads();
      if (tid == 0) act[jrow + py * P_ + px] = red[0] + red[1] + red[2] + red[3];
      __syncthreads();
    }
  }
}

// ---------------- state update + output write ----------------
__global__ void k_update(const float* __restrict__ resbuf, float* __restrict__ traj,
                         float* __restrict__ qv, float* __restrict__ out) {
  int idx = blockIdx.x * blockDim.x + threadIdx.x;
  if (idx >= T_ * B_ * N_) return;
  int t = idx / (B_ * N_); int r = idx % (B_ * N_);
  int b = r / N_; int n = r % N_;
  int tb = t * B_ + b;
  size_t j = (size_t)(b * N_ + n) * T_ + t;
  const float* rr = resbuf + j * CP_;
  size_t tp = ((size_t)tb * N_ + n) * 2;
  float nx = traj[tp + 0] + rr[0];
  float ny = traj[tp + 1] + rr[1];
  traj[tp + 0] = nx; traj[tp + 1] = ny;
  float* qp = qv + ((size_t)tb * N_ + n) * C_;
  for (int c = 0; c < C_; ++c) qp[c] += rr[2 + c];
  // outs: [T,B,2,N] scaled by 4
  out[((size_t)tb * 2 + 0) * N_ + n] = nx * 4.f;
  out[((size_t)tb * 2 + 1) * N_ + n] = ny * 4.f;
}

// =============================== host launcher ===============================
static inline dim3 gblk(long long n, int bs) { return dim3((unsigned)((n + bs - 1) / bs)); }

extern "C" void kernel_launch(void* const* d_in, const int* in_sizes, int n_in,
                              void* d_out, int out_size, void* d_ws, size_t ws_size,
                              hipStream_t stream) {
  (void)in_sizes; (void)n_in; (void)out_size; (void)ws_size;
  // Input flatten order: traj_init, query_vectors_init, features, then params pytree
  // (dict keys sorted: blocks[0..5]{c1_b,c1_w,dw_b,dw_w,pw_b,pw_w}, in_b, in_w, out_b, out_w)
  const float* traj_init = (const float*)d_in[0];
  const float* qv_init   = (const float*)d_in[1];
  const float* features  = (const float*)d_in[2];
  const float *c1_b[6], *c1_w[6], *dw_b[6], *dw_w[6], *pw_b[6], *pw_w[6];
  for (int i = 0; i < 6; ++i) {
    int base = 3 + 6 * i;
    c1_b[i] = (const float*)d_in[base + 0];
    c1_w[i] = (const float*)d_in[base + 1];
    dw_b[i] = (const float*)d_in[base + 2];
    dw_w[i] = (const float*)d_in[base + 3];
    pw_b[i] = (const float*)d_in[base + 4];
    pw_w[i] = (const float*)d_in[base + 5];
  }
  const float* in_b_  = (const float*)d_in[39];
  const float* in_w_  = (const float*)d_in[40];
  const float* out_b_ = (const float*)d_in[41];
  const float* out_w_ = (const float*)d_in[42];

  // ---- carve workspace ----
  char* p = (char*)d_ws;
  auto carve = [&](size_t bytes) -> void* {
    void* r = (void*)p;
    p += (bytes + 255) & ~(size_t)255;
    return r;
  };
  float* traj = (float*)carve((size_t)T_ * B_ * N_ * 2 * 4);
  float* qv   = (float*)carve((size_t)T_ * B_ * N_ * C_ * 4);
  float* pyr1 = (float*)carve((size_t)T_ * B_ * C_ * 48 * 48 * 4);
  float* pyr2 = (float*)carve((size_t)T_ * B_ * C_ * 24 * 24 * 4);
  float* pyr3 = (float*)carve((size_t)T_ * B_ * C_ * 12 * 12 * 4);
  float* actX = (float*)carve((size_t)NJ_ * CP_ * 4);
  float* actY = (float*)carve((size_t)NJ_ * CP_ * 4);
  float* actT = (float*)carve((size_t)NJ_ * CP_ * 4);
  _Float16* xh5    = (_Float16*)carve(5 * XSLOT_ * 2);   // 5 shifted slots; slot 0 = 1x1 path
  _Float16* wh_in  = (_Float16*)carve((size_t)5 * MP_MAIN * KP_ * 2);
  _Float16* wh_c1[6]; _Float16* wh_pw[6];
  for (int i = 0; i < 6; ++i) {
    wh_c1[i] = (_Float16*)carve((size_t)MP_MAIN * KP_ * 2);
    wh_pw[i] = (_Float16*)carve((size_t)MP_MAIN * KP_ * 2);
  }
  _Float16* wh_out = (_Float16*)carve((size_t)MP_OUT * KP_ * 2);

  // ---- state init ----
  k_init_traj<<<gblk(T_ * B_ * N_, 256), 256, 0, stream>>>(traj_init, traj);
  k_init_qv<<<gblk((long long)T_ * B_ * N_ * C_, 256), 256, 0, stream>>>(qv_init, qv);

  // ---- feature pyramid ----
  k_pool<<<gblk((long long)T_ * B_ * C_ * 48 * 48, 256), 256, 0, stream>>>(features, pyr1, 96, 96);
  k_pool<<<gblk((long long)T_ * B_ * C_ * 24 * 24, 256), 256, 0, stream>>>(pyr1, pyr2, 48, 48);
  k_pool<<<gblk((long long)T_ * B_ * C_ * 12 * 12, 256), 256, 0, stream>>>(pyr2, pyr3, 24, 24);

  // ---- weights -> f16 padded tiles ----
  k_wcast<<<gblk((long long)5 * MP_MAIN * KP_, 256), 256, 0, stream>>>(in_w_, wh_in, CIN_, CIN_, 5, MP_MAIN);
  for (int i = 0; i < 6; ++i) {
    k_wcast<<<gblk((long long)MP_MAIN * KP_, 256), 256, 0, stream>>>(c1_w[i], wh_c1[i], CIN_, CIN_, 1, MP_MAIN);
    k_wcast<<<gblk((long long)MP_MAIN * KP_, 256), 256, 0, stream>>>(pw_w[i], wh_pw[i], CIN_, CIN_, 1, MP_MAIN);
  }
  k_wcast<<<gblk((long long)MP_OUT * KP_, 256), 256, 0, stream>>>(out_w_, wh_out, COUT_, CIN_, 1, MP_OUT);

  const float* pyr_ptr[4] = {features, pyr1, pyr2, pyr3};
  const int pyr_hw[4] = {96, 48, 24, 12};

  const int Mtiles_main = MP_MAIN / 16;  // 25
  const int Mtiles_out  = MP_OUT / 16;   // 9
  const long long stage_n = (long long)KP_ * NJ_;
  auto gemm_blocks = [](int mtiles) {
    return dim3((unsigned)((mtiles * (NJ_ / (16 * NT_)) + 7) / 8));
  };

  for (int it = 0; it < ITERS_; ++it) {
    // base channels + correlation channels into actX
    k_build<<<gblk(T_ * B_ * N_, 256), 256, 0, stream>>>(traj, qv, actX);
    for (int lvl = 0; lvl < 4; ++lvl) {
      float inv = 1.f / (float)(1 << lvl);
      k_corr<<<dim3(T_ * B_ * N_), dim3(128), 0, stream>>>(
          pyr_ptr[lvl], pyr_hw[lvl], pyr_hw[lvl], traj, qv, actX, inv,
          4 + EMB_ + C_ + lvl * P_ * P_);
    }

    // basenet input conv (k=5): 5 pre-shifted stagings -> 5-tap GEMM, then inorm+gelu
    for (int s = 0; s < 5; ++s)
      k_stage<<<gblk(stage_n, 256), 256, 0, stream>>>(actX, xh5 + (size_t)s * XSLOT_, s - 2);
    k_gemm_wmma<<<gemm_blocks(Mtiles_main), 256, 0, stream>>>(xh5, wh_in, in_b_, actY,
                                                              CIN_, Mtiles_main, MP_MAIN, 5);
    k_inorm_gelu<<<gblk((long long)BN_ * CIN_, 256), 256, 0, stream>>>(actY, nullptr, actX);

    for (int blk = 0; blk < 6; ++blk) {
      // x1 = gelu(inorm(conv1(x)) + x)
      k_stage<<<gblk(stage_n, 256), 256, 0, stream>>>(actX, xh5, 0);
      k_gemm_wmma<<<gemm_blocks(Mtiles_main), 256, 0, stream>>>(xh5, wh_c1[blk], c1_b[blk], actY,
                                                                CIN_, Mtiles_main, MP_MAIN, 1);
      k_inorm_gelu<<<gblk((long long)BN_ * CIN_, 256), 256, 0, stream>>>(actY, actX, actT);
      // y = pw(dw(x1)) ; x = gelu(inorm(y) + x1)
      k_dw<<<gblk((long long)BN_ * CIN_, 256), 256, 0, stream>>>(actT, dw_w[blk], dw_b[blk], actY);
      k_stage<<<gblk(stage_n, 256), 256, 0, stream>>>(actY, xh5, 0);
      k_gemm_wmma<<<gemm_blocks(Mtiles_main), 256, 0, stream>>>(xh5, wh_pw[blk], pw_b[blk], actY,
                                                                CIN_, Mtiles_main, MP_MAIN, 1);
      k_inorm_gelu<<<gblk((long long)BN_ * CIN_, 256), 256, 0, stream>>>(actY, actT, actX);
    }

    // output conv (392 -> 130) + state update + write outs[it]
    k_stage<<<gblk(stage_n, 256), 256, 0, stream>>>(actX, xh5, 0);
    k_gemm_wmma<<<gemm_blocks(Mtiles_out), 256, 0, stream>>>(xh5, wh_out, out_b_, actY,
                                                             COUT_, Mtiles_out, MP_OUT, 1);
    k_update<<<gblk(T_ * B_ * N_, 256), 256, 0, stream>>>(
        actY, traj, qv, (float*)d_out + (size_t)it * T_ * B_ * 2 * N_);
  }
}
